// SAttention_16793322127944
// MI455X (gfx1250) — matrix operand order repaired and checked by reading, not compiled, not run on hardware
//
#include <hip/hip_runtime.h>

// ---------------------------------------------------------------------------
// Transformer encoder layer for MI455X (gfx1250, wave32, WMMA + TDM).
// S=2048, B=4, D=512, H=8, HD=64, M = S*B = 8192.
// bf16 WMMA for all GEMMs + flash attention; weights staged into LDS via the
// Tensor Data Mover (tensor_load_to_lds) with double buffering.
// ---------------------------------------------------------------------------

#define S_DIM 2048
#define B_DIM 4
#define D_DIM 512
#define H_DIM 8
#define HD_DIM 64
#define M_DIM (S_DIM * B_DIM)   // 8192
#define INV_TEMP 0.125f         // 1/sqrt(HD)

typedef __attribute__((ext_vector_type(16))) __bf16 v16bf;
typedef __attribute__((ext_vector_type(8)))  __bf16 v8bf;
typedef __attribute__((ext_vector_type(8)))  float  v8f;
typedef __attribute__((ext_vector_type(4)))  unsigned int u32x4;
typedef __attribute__((ext_vector_type(8)))  int i32x8;
typedef __attribute__((ext_vector_type(4)))  int i32x4;

// ---------------------------------------------------------------------------
// helpers
// ---------------------------------------------------------------------------
__device__ __forceinline__ v8f wmma_bf16(v16bf a, v16bf b, v8f c) {
    // D = A(16x32 bf16) * B(32x16 bf16) + C(16x16 f32)
    return __builtin_amdgcn_wmma_f32_16x16x32_bf16(
        /*neg_a=*/false, a, /*neg_b=*/false, b,
        /*c_mod=*/(short)0, c, /*reuse_a=*/false, /*reuse_b=*/false);
}

__device__ __forceinline__ float redmax16(float v) {
    v = fmaxf(v, __shfl_xor(v, 1, 32));
    v = fmaxf(v, __shfl_xor(v, 2, 32));
    v = fmaxf(v, __shfl_xor(v, 4, 32));
    v = fmaxf(v, __shfl_xor(v, 8, 32));
    return v;
}
__device__ __forceinline__ float redsum16(float v) {
    v += __shfl_xor(v, 1, 32);
    v += __shfl_xor(v, 2, 32);
    v += __shfl_xor(v, 4, 32);
    v += __shfl_xor(v, 8, 32);
    return v;
}
__device__ __forceinline__ float redsum32(float v) {
    v += __shfl_xor(v, 1, 32);
    v += __shfl_xor(v, 2, 32);
    v += __shfl_xor(v, 4, 32);
    v += __shfl_xor(v, 8, 32);
    v += __shfl_xor(v, 16, 32);
    return v;
}

// A-fragment (16x32 bf16, row major source).
// Lanes 0-15: row=lane, K={k0..k0+15};
// lanes 16-31: row=lane-16, K={k0+16..k0+31}.
__device__ __forceinline__ v16bf load_afrag(const __bf16* rowptr, int k0, int lane) {
    const int b0 = k0 + ((lane & 16) ? 16 : 0);
    v8bf lo = *(const v8bf*)(rowptr + b0);
    v8bf hi = *(const v8bf*)(rowptr + b0 + 8);
    union { v16bf v; v8bf h[2]; } u;
    u.h[0] = lo; u.h[1] = hi;
    return u.v;
}

// ---------------------------------------------------------------------------
// TDM: 2D tile load (global -> LDS), bf16 elements (data_size = 2 bytes).
// D# layout per CDNA5 ISA ch. 8 (group0 128b, group1 256b; groups 2/3 zero).
// lds_off is a byte offset into the workgroup's static LDS (base 0).
// 6-arg builtin form (clang-23 / therock-10.0 toolchain).
// ---------------------------------------------------------------------------
__device__ __forceinline__ void tdm_load_tile_2d(unsigned lds_off, const void* gptr,
                                                 unsigned tile0, unsigned tile1,
                                                 unsigned tdim0, unsigned tdim1,
                                                 unsigned long long stride0) {
    unsigned long long ga = (unsigned long long)(uintptr_t)gptr;
    u32x4 g0;
    g0[0] = 1u;                                            // count=1, user mode
    g0[1] = lds_off;                                       // lds_addr (bytes)
    g0[2] = (unsigned)(ga & 0xFFFFFFFFull);                // global_addr[31:0]
    g0[3] = (unsigned)((ga >> 32) & 0x01FFFFFFull)         // global_addr[56:32]
            | (2u << 30);                                  // type = 2 ("image")
    i32x8 g1 = {};
    g1[0] = (int)(1u << 16);                               // data_size=1 (2 bytes)
    g1[1] = (int)((tdim0 & 0xFFFFu) << 16);                // tensor_dim0[15:0]
    g1[2] = (int)(((tdim0 >> 16) & 0xFFFFu) |
                  ((tdim1 & 0xFFFFu) << 16));              // dim0 hi | dim1 lo
    g1[3] = (int)(((tdim1 >> 16) & 0xFFFFu) |
                  ((tile0 & 0xFFFFu) << 16));              // dim1 hi | tile_dim0
    g1[4] = (int)(tile1 & 0xFFFFu);                        // tile_dim1 (tile_dim2=0)
    g1[5] = (int)(unsigned)(stride0 & 0xFFFFFFFFull);      // tensor_dim0_stride lo
    g1[6] = (int)(unsigned)((stride0 >> 32) & 0xFFFFull);  // stride hi (dim1_stride=0)
    g1[7] = 0;
    i32x4 z4 = {};
    i32x8 z8 = {};
    __builtin_amdgcn_tensor_load_to_lds(g0, g1, z4, z4, z8, 0);
}

// ---------------------------------------------------------------------------
// Weight transpose + cast: W[K,N] f32 -> Wt[N,K] bf16   (K = N = 512)
// ---------------------------------------------------------------------------
__global__ __launch_bounds__(256) void transpose_cast_kernel(const float* __restrict__ w,
                                                             __bf16* __restrict__ wt) {
    int idx = blockIdx.x * 256 + threadIdx.x;          // 0 .. 512*512-1
    int n = idx >> 9;
    int k = idx & 511;
    wt[(size_t)n * D_DIM + k] = (__bf16)w[(size_t)k * D_DIM + n];
}

// ---------------------------------------------------------------------------
// (optional residual add) + LayerNorm. One wave per row (D=512, 16 elems/lane).
// ---------------------------------------------------------------------------
__global__ __launch_bounds__(256) void add_ln_kernel(const float* __restrict__ x,
                                                     const float* __restrict__ res,
                                                     const float* __restrict__ g,
                                                     const float* __restrict__ beta,
                                                     float* __restrict__ xt_out,
                                                     __bf16* __restrict__ xn_out) {
    const int lane = threadIdx.x & 31;
    const int wave = threadIdx.x >> 5;
    const int row = blockIdx.x * 8 + wave;             // 8 rows per block
    const float* xr = x + (size_t)row * D_DIM;
    const float* rr = res ? res + (size_t)row * D_DIM : nullptr;

    float vals[16];
    float acc = 0.f;
#pragma unroll
    for (int i = 0; i < 16; ++i) {
        int c = lane + 32 * i;
        float v = xr[c];
        if (rr) v += rr[c];
        vals[i] = v;
        acc += v;
    }
    float mu = redsum32(acc) * (1.0f / D_DIM);
    float vacc = 0.f;
#pragma unroll
    for (int i = 0; i < 16; ++i) {
        float d = vals[i] - mu;
        vacc += d * d;
    }
    float rstd = rsqrtf(redsum32(vacc) * (1.0f / D_DIM) + 1e-5f);

#pragma unroll
    for (int i = 0; i < 16; ++i) {
        int c = lane + 32 * i;
        float y = (vals[i] - mu) * rstd * g[c] + beta[c];
        xn_out[(size_t)row * D_DIM + c] = (__bf16)y;
        if (xt_out) xt_out[(size_t)row * D_DIM + c] = vals[i];
    }
}

// ---------------------------------------------------------------------------
// GEMM: C[M=8192, N=512] = A[M,512] (bf16 row-major) x Wt[N,512] (bf16 = W^T)
// Block = 4 waves = 64(M) x 64(N). Weight tile (64 N-rows x 128 K) staged into
// LDS by the Tensor Data Mover, double-buffered; wave 0 drives the TDM and
// waits on TENSORcnt, barriers publish to the other waves. B-fragments are
// then read from LDS; A-fragments load directly from global.
// MODE 0: bf16 -> q/k layout [B,H,S,HD];  MODE 1: bf16 -> v^T layout [B,H,HD,S]
// MODE 2: bf16 relu(acc+bias) [M,N];      MODE 3: f32 acc+bias+resid [M,N]
// ---------------------------------------------------------------------------
#define GK_CHUNK 128                      // K elements staged per TDM transfer
#define GK_BUFBYTES (64 * GK_CHUNK * 2)   // 16 KB per buffer

template <int MODE>
__global__ __launch_bounds__(128) void gemm_kernel(const __bf16* __restrict__ A,
                                                   const __bf16* __restrict__ Wt,
                                                   void* __restrict__ out,
                                                   const float* __restrict__ bias,
                                                   const float* __restrict__ resid) {
    __shared__ __align__(64) __bf16 bsm[2][64 * GK_CHUNK];   // 32 KB double buffer

    const int lane  = threadIdx.x & 31;
    const int wave  = threadIdx.x >> 5;
    const int mt64  = blockIdx.x >> 3;        // 0..127 (64-row M block)
    const int ntile = blockIdx.x & 7;         // 0..7   (64-col N block)
    const int nl    = lane & 15;
    const int hb    = (lane >> 4) & 1;

    const int arow = mt64 * 64 + wave * 16 + nl;
    const __bf16* arowp = A + (size_t)arow * D_DIM;
    const __bf16* wtile = Wt + (size_t)ntile * 64 * D_DIM;   // 64 rows x 512

    // Kick off chunk 0 via TDM (wave 0 only; EXEC is ignored by TDM).
    if (wave == 0) {
        tdm_load_tile_2d(/*lds_off=*/0u, wtile,
                         /*tile0=*/GK_CHUNK, /*tile1=*/64,
                         /*tdim0=*/D_DIM, /*tdim1=*/64, /*stride0=*/D_DIM);
    }

    v8f acc[4] = {};

    for (int chunk = 0; chunk < D_DIM / GK_CHUNK; ++chunk) {
        if (wave == 0) __builtin_amdgcn_s_wait_tensorcnt(0);
        __syncthreads();   // chunk's data visible to all waves
        // overlap: start DMA of the next chunk into the other buffer
        if (wave == 0 && chunk + 1 < D_DIM / GK_CHUNK) {
            tdm_load_tile_2d((unsigned)(((chunk + 1) & 1) * GK_BUFBYTES),
                             wtile + (chunk + 1) * GK_CHUNK,
                             GK_CHUNK, 64, D_DIM, 64, D_DIM);
        }
        const __bf16* bbuf = &bsm[chunk & 1][0];
#pragma unroll
        for (int ks = 0; ks < GK_CHUNK / 32; ++ks) {
            const int k0 = chunk * GK_CHUNK + ks * 32;
            v16bf af = load_afrag(arowp, k0, lane);
#pragma unroll
            for (int c = 0; c < 4; ++c) {
                // LDS tile layout: [local n row][GK_CHUNK k], rows contiguous
                v16bf bf = *(const v16bf*)(bbuf + (size_t)(c * 16 + nl) * GK_CHUNK +
                                           ks * 32 + hb * 16);
                acc[c] = wmma_bf16(af, bf, acc[c]);
            }
        }
        __syncthreads();   // all waves done reading before buffer is re-filled
    }

#pragma unroll
    for (int c = 0; c < 4; ++c) {
        const int col = ntile * 64 + c * 16 + nl;
        float bv = 0.f;
        if (MODE == 2 || MODE == 3) bv = bias[col];
#pragma unroll
        for (int r = 0; r < 8; ++r) {
            const int m = mt64 * 64 + wave * 16 + r + 8 * hb;
            float v = acc[c][r];
            if (MODE == 0) {
                int s = m >> 2, b = m & 3;                 // m = s*B + b
                int h = col >> 6, hd = col & 63;
                ((__bf16*)out)[(((size_t)(b * H_DIM + h) * S_DIM + s) * HD_DIM) + hd] = (__bf16)v;
            } else if (MODE == 1) {
                int s = m >> 2, b = m & 3;
                int h = col >> 6, hd = col & 63;
                ((__bf16*)out)[(((size_t)(b * H_DIM + h) * HD_DIM + hd) * S_DIM) + s] = (__bf16)v;
            } else if (MODE == 2) {
                ((__bf16*)out)[(size_t)m * D_DIM + col] = (__bf16)fmaxf(v + bv, 0.f);
            } else {
                ((float*)out)[(size_t)m * D_DIM + col] =
                    v + bv + resid[(size_t)m * D_DIM + col];
            }
        }
    }
}

// ---------------------------------------------------------------------------
// Flash attention. One wave handles (b,h) + a 16-row Q tile; streams 32-key
// blocks with online softmax. P-fragment relayout via per-wave LDS tile.
// q,k: [B,H,S,HD] bf16; vt: [B,H,HD,S] bf16; att out: [S,B,D] f32.
// ---------------------------------------------------------------------------
__global__ __launch_bounds__(128) void flash_attn_kernel(const __bf16* __restrict__ q,
                                                         const __bf16* __restrict__ k,
                                                         const __bf16* __restrict__ vt,
                                                         float* __restrict__ att) {
    __shared__ float sm[4][16 * 33];      // per-wave 16x32 P tile (+pad)

    const int lane = threadIdx.x & 31;
    const int wave = threadIdx.x >> 5;
    const int wid  = blockIdx.x * 4 + wave;     // 0..4095
    const int bh   = wid >> 7;                  // 0..31
    const int qt   = wid & 127;                 // 0..127 (Q row tiles)
    const int b    = bh >> 3;
    const int h    = bh & 7;

    const __bf16* qb = q  + (size_t)bh * S_DIM * HD_DIM;
    const __bf16* kb = k  + (size_t)bh * S_DIM * HD_DIM;
    const __bf16* vb = vt + (size_t)bh * HD_DIM * S_DIM;
    float* smw = &sm[wave][0];

    const int nl = lane & 15;
    const int hb = (lane >> 4) & 1;

    // preload Q A-fragments for the two 32-wide HD chunks
    const int qrow = qt * 16 + nl;
    v16bf qa[2];
#pragma unroll
    for (int kc = 0; kc < 2; ++kc)
        qa[kc] = load_afrag(qb + (size_t)qrow * HD_DIM, kc * 32, lane);

    float mrow[8], lrow[8];
#pragma unroll
    for (int r = 0; r < 8; ++r) { mrow[r] = -1e30f; lrow[r] = 0.f; }
    v8f o[4] = {};

    for (int kb0 = 0; kb0 < S_DIM; kb0 += 32) {
        // pull the next key/value block toward the WGP (global_prefetch_b8)
        if (kb0 + 32 < S_DIM) {
            __builtin_prefetch(kb + (size_t)(kb0 + 32 + lane) * HD_DIM, 0, 0);
            __builtin_prefetch(vb + (size_t)(lane << 1) * S_DIM + kb0 + 32, 0, 0);
        }

        // ---- scores for 32 keys (two 16-key subtiles) ----
        v8f s0 = {}, s1 = {};
        {
            int key0 = kb0 + nl;
            v16bf kf0a = *(const v16bf*)(kb + (size_t)key0 * HD_DIM + hb * 16);
            v16bf kf0b = *(const v16bf*)(kb + (size_t)key0 * HD_DIM + 32 + hb * 16);
            s0 = wmma_bf16(qa[0], kf0a, s0);
            s0 = wmma_bf16(qa[1], kf0b, s0);
            int key1 = kb0 + 16 + nl;
            v16bf kf1a = *(const v16bf*)(kb + (size_t)key1 * HD_DIM + hb * 16);
            v16bf kf1b = *(const v16bf*)(kb + (size_t)key1 * HD_DIM + 32 + hb * 16);
            s1 = wmma_bf16(qa[0], kf1a, s1);
            s1 = wmma_bf16(qa[1], kf1b, s1);
        }

        // ---- online softmax update + stage P into LDS ----
#pragma unroll
        for (int r = 0; r < 8; ++r) {
            float x0 = s0[r] * INV_TEMP;
            float x1 = s1[r] * INV_TEMP;
            float rm = redmax16(fmaxf(x0, x1));
            float mn = fmaxf(mrow[r], rm);
            float alpha = __expf(mrow[r] - mn);
            float p0 = __expf(x0 - mn);
            float p1 = __expf(x1 - mn);
            float rs = redsum16(p0 + p1);
            lrow[r] = lrow[r] * alpha + rs;
            mrow[r] = mn;
            o[0][r] *= alpha; o[1][r] *= alpha; o[2][r] *= alpha; o[3][r] *= alpha;
            int rr = r + 8 * hb;
            smw[rr * 33 + nl]      = p0;
            smw[rr * 33 + 16 + nl] = p1;
        }
        // all lanes' ds_stores must land before cross-lane ds_loads
        asm volatile("s_wait_dscnt 0x0" ::: "memory");

        // ---- rebuild P as a 16x32 bf16 A-fragment ----
        v16bf pf;
        {
            const int cb0 = hb ? 16 : 0;
#pragma unroll
            for (int i = 0; i < 8; ++i)
                pf[i] = (__bf16)smw[nl * 33 + cb0 + i];
#pragma unroll
            for (int i = 0; i < 8; ++i)
                pf[8 + i] = (__bf16)smw[nl * 33 + cb0 + 8 + i];
        }

        // ---- O += P(16x32) * V(32x64) ----
#pragma unroll
        for (int c = 0; c < 4; ++c) {
            int hd = c * 16 + nl;
            v16bf vf = *(const v16bf*)(vb + (size_t)hd * S_DIM + kb0 + hb * 16);
            o[c] = wmma_bf16(pf, vf, o[c]);
        }
    }

    // ---- normalize and write out: att[s, b, h*64+hd] ----
#pragma unroll
    for (int c = 0; c < 4; ++c) {
#pragma unroll
        for (int r = 0; r < 8; ++r) {
            int srow = qt * 16 + r + 8 * hb;
            float v = o[c][r] / lrow[r];
            att[((size_t)srow * B_DIM + b) * D_DIM + h * HD_DIM + c * 16 + nl] = v;
        }
    }
}

// ---------------------------------------------------------------------------
// Host-side orchestration (all on `stream`, graph-capture safe).
// ---------------------------------------------------------------------------
extern "C" void kernel_launch(void* const* d_in, const int* in_sizes, int n_in,
                              void* d_out, int out_size, void* d_ws, size_t ws_size,
                              hipStream_t stream) {
    (void)in_sizes; (void)n_in; (void)out_size; (void)ws_size;

    const float* x     = (const float*)d_in[0];
    const float* w_q   = (const float*)d_in[1];
    const float* w_k   = (const float*)d_in[2];
    const float* w_v   = (const float*)d_in[3];
    const float* g1    = (const float*)d_in[4];
    const float* beta1 = (const float*)d_in[5];
    const float* g2    = (const float*)d_in[6];
    const float* beta2 = (const float*)d_in[7];
    const float* w1    = (const float*)d_in[8];
    const float* bias1 = (const float*)d_in[9];
    const float* w2    = (const float*)d_in[10];
    const float* bias2 = (const float*)d_in[11];
    float* out = (float*)d_out;

    // ---- workspace layout (bytes) ----
    char* ws = (char*)d_ws;
    const size_t MB = (size_t)M_DIM * D_DIM;       // 8192*512 elements
    size_t off = 0;
    __bf16* xn  = (__bf16*)(ws + off); off += MB * 2;            // 8 MB
    __bf16* qb  = (__bf16*)(ws + off); off += MB * 2;            // 8 MB
    __bf16* kbf = (__bf16*)(ws + off); off += MB * 2;            // 8 MB
    __bf16* vtb = (__bf16*)(ws + off); off += MB * 2;            // 8 MB
    float*  att = (float*) (ws + off); off += MB * 4;            // 16 MB
    float*  xt  = (float*) (ws + off); off += MB * 4;            // 16 MB
    __bf16* wqT = (__bf16*)(ws + off); off += (size_t)D_DIM * D_DIM * 2;
    __bf16* wkT = (__bf16*)(ws + off); off += (size_t)D_DIM * D_DIM * 2;
    __bf16* wvT = (__bf16*)(ws + off); off += (size_t)D_DIM * D_DIM * 2;
    __bf16* w1T = (__bf16*)(ws + off); off += (size_t)D_DIM * D_DIM * 2;
    __bf16* w2T = (__bf16*)(ws + off); off += (size_t)D_DIM * D_DIM * 2;
    // aliases for post-attention stages (xn / q dead by then)
    __bf16* xtn = xn;    // LN2 output (bf16)
    __bf16* h1  = qb;    // FFN hidden (bf16)

    const dim3 blk256(256), blk128(128);

    // 1) weight transposes: 512*512 / 256 = 1024 blocks each
    transpose_cast_kernel<<<1024, blk256, 0, stream>>>(w_q, wqT);
    transpose_cast_kernel<<<1024, blk256, 0, stream>>>(w_k, wkT);
    transpose_cast_kernel<<<1024, blk256, 0, stream>>>(w_v, wvT);
    transpose_cast_kernel<<<1024, blk256, 0, stream>>>(w1, w1T);
    transpose_cast_kernel<<<1024, blk256, 0, stream>>>(w2, w2T);

    // 2) LN1 -> bf16 (8192 rows, 8 rows/block)
    add_ln_kernel<<<M_DIM / 8, blk256, 0, stream>>>(x, nullptr, g1, beta1, nullptr, xn);

    // 3) QKV projections: 1024 blocks of 128 (4 waves = 64x64 tile per block)
    gemm_kernel<0><<<1024, blk128, 0, stream>>>(xn, wqT, qb, nullptr, nullptr);
    gemm_kernel<0><<<1024, blk128, 0, stream>>>(xn, wkT, kbf, nullptr, nullptr);
    gemm_kernel<1><<<1024, blk128, 0, stream>>>(xn, wvT, vtb, nullptr, nullptr);

    // 4) flash attention: 32 (b,h) * 128 q-tiles = 4096 waves -> 1024 blocks
    flash_attn_kernel<<<1024, blk128, 0, stream>>>(qb, kbf, vtb, att);

    // 5) residual + LN2 (keeps f32 xt for final residual, bf16 xtn for FFN)
    add_ln_kernel<<<M_DIM / 8, blk256, 0, stream>>>(x, att, g2, beta2, xt, xtn);

    // 6) FFN
    gemm_kernel<2><<<1024, blk128, 0, stream>>>(xtn, w1T, h1, bias1, nullptr);
    gemm_kernel<3><<<1024, blk128, 0, stream>>>(h1, w2T, out, bias2, xt);
}
